// TimeOfFlightNet_70394513981864
// MI455X (gfx1250) — matrix operand order repaired and checked by reading, not compiled, not run on hardware
//
#include <hip/hip_runtime.h>
#include <hip/hip_bf16.h>

typedef __attribute__((ext_vector_type(16))) _Float16 v16h;
typedef __attribute__((ext_vector_type(8)))  float    v8f;

#define REC_L     4096
#define CROP_K    128
#define NRECV     8
#define NB2       16384
#define HIDDEN    32
#define KDIM      (NRECV * CROP_K)   // 1024
#define IX_STEP   (0.25f / (float)REC_L)

// fused: distances -> hat weights -> implicit A tile -> WMMA GEMM -> MLP head
__global__ __launch_bounds__(256) void tof_wmma_kernel(
    const float* __restrict__ recordings,       // (4, 8, 4096)
    const float* __restrict__ sample_locations, // (4, 16384, 3)
    const float* __restrict__ emitter,          // (3,)
    const float* __restrict__ receivers,        // (8, 3)
    const float* __restrict__ W1,               // (1024, 32)
    const float* __restrict__ b1,               // (32,)
    const float* __restrict__ W2,               // (32, 1)
    const float* __restrict__ b2,               // (1,)
    float* __restrict__ out)                    // (4, 16384)
{
    // W1 pre-scaled by vy[b1][r], as f16 in LDS, row-major (1024 x 32): 64 KB.
    // vy is block-uniform: each block covers 128 consecutive rows and
    // 16384 % 128 == 0, so b1 = blockIdx.x >> 7 for the whole block.
    __shared__ _Float16 W1h[KDIM * HIDDEN];

    const int tid   = threadIdx.x;
    const int b1blk = blockIdx.x >> 7;

    #pragma unroll
    for (int r = 0; r < NRECV; ++r) {
        // center-tap blend of the recording (uniform address -> scalar loads)
        const float* rec = &recordings[(size_t)(b1blk * NRECV + r) * REC_L];
        const float vyr = 0.5f * (rec[2047] + rec[2048]);
        const int base = r * CROP_K * HIDDEN;
        for (int i = tid; i < CROP_K * HIDDEN; i += 256)
            W1h[base + i] = (_Float16)(W1[base + i] * vyr);
    }
    __syncthreads();

    const int lane = tid & 31;
    const int wave = tid >> 5;
    const int half = lane >> 4;    // which K-half of the A layout this lane holds
    const int m    = lane & 15;    // matrix row within the tile (2 lanes per row)

    const int tile = blockIdx.x * 8 + wave;   // 512 blocks * 8 waves = 4096 tiles
    const int n    = tile * 16 + m;           // global sample row

    __builtin_prefetch(&sample_locations[(size_t)n * 3], 0, 0);

    // emitter distance for this row
    const float ex = emitter[0], ey = emitter[1], ez = emitter[2];
    const float sx = sample_locations[n * 3 + 0];
    const float sy = sample_locations[n * 3 + 1];
    const float sz = sample_locations[n * 3 + 2];
    const float d_emit = sqrtf((sx - ex) * (sx - ex) +
                               (sy - ey) * (sy - ey) +
                               (sz - ez) * (sz - ez));

    // runtime part of the A-layout K index, folded once (not per element!)
    const float ix_half = (float)(half * 8) * IX_STEP;

    // 4 independent accumulator chains (even/odd chunk) to hide the
    // f16-WMMA hazard window instead of v_nop padding.
    v8f acc0a = {}, acc0b = {};   // hidden units 0..15
    v8f acc1a = {}, acc1b = {};   // hidden units 16..31

    #pragma unroll
    for (int r = 0; r < NRECV; ++r) {
        const float rx = receivers[r * 3 + 0];
        const float ry = receivers[r * 3 + 1];
        const float rz = receivers[r * 3 + 2];
        const float d_recv = sqrtf((sx - rx) * (sx - rx) +
                                   (sy - ry) * (sy - ry) +
                                   (sz - rz) * (sz - rz));

        // crop_start = round(tof_samples) - K/2 ; RNE matches jnp.round.
        const float crop_start =
            rintf((d_emit + d_recv) * (96000.0f / 343.0f)) - 64.0f;
        // ix(k) = 0.5*gx(k) = (-0.5 + crop_start/L) + k * 0.25/L  (0.5 folded in)
        const float ixlane = (-0.5f + crop_start * (1.0f / (float)REC_L)) + ix_half;

        #pragma unroll
        for (int c = 0; c < CROP_K / 32; ++c) {       // 4 chunks of K=32
            // Build A tile (16x32 f16) directly in WMMA register layout:
            // lane holds row m; element e maps to K = c*32 + (e<8?0:16) + half*8 + (e&7).
            // half*8 is folded into ixlane; the rest is a compile-time literal.
            v16h a;
            #pragma unroll
            for (int e = 0; e < 16; ++e) {
                const float koff =
                    (float)(c * 32 + ((e < 8) ? 0 : 16) + (e & 7)) * IX_STEP;
                const float ix = ixlane + koff;                  // v_add literal
                const float w  = fmaxf(0.0f, 1.0f - fabsf(ix));  // exact hat
                a[e] = (_Float16)w;                   // vy already baked into B
            }

            // B tiles (32x16 f16): lane holds column N = m; element e maps to same K as A
            v16h bA, bB;
            #pragma unroll
            for (int e = 0; e < 16; ++e) {
                const int krow = r * CROP_K + c * 32 + ((e < 8) ? 0 : 16) + half * 8 + (e & 7);
                bA[e] = W1h[krow * HIDDEN + m];
                bB[e] = W1h[krow * HIDDEN + 16 + m];
            }

            if (c & 1) {
                acc0b = __builtin_amdgcn_wmma_f32_16x16x32_f16(
                    false, a, false, bA, (short)0, acc0b, false, false);
                acc1b = __builtin_amdgcn_wmma_f32_16x16x32_f16(
                    false, a, false, bB, (short)0, acc1b, false, false);
            } else {
                acc0a = __builtin_amdgcn_wmma_f32_16x16x32_f16(
                    false, a, false, bA, (short)0, acc0a, false, false);
                acc1a = __builtin_amdgcn_wmma_f32_16x16x32_f16(
                    false, a, false, bB, (short)0, acc1a, false, false);
            }
        }
    }

    // MLP head: relu(h + b1) @ W2 + b2, reduced across the 16 N-lanes.
    // acc element i lives at (M = i + 8*half, N = m).
    const float b1A = b1[m],      b1B = b1[16 + m];
    const float w2A = W2[m],      w2B = W2[16 + m];
    const float b2v = b2[0];

    #pragma unroll
    for (int i = 0; i < 8; ++i) {
        const float h0 = acc0a[i] + acc0b[i];
        const float h1 = acc1a[i] + acc1b[i];
        float s = fmaxf(h0 + b1A, 0.0f) * w2A +
                  fmaxf(h1 + b1B, 0.0f) * w2B;
        // reduce over the 16 lanes of this half (masks 1..8 stay in-half)
        s += __shfl_xor(s, 8, 32);
        s += __shfl_xor(s, 4, 32);
        s += __shfl_xor(s, 2, 32);
        s += __shfl_xor(s, 1, 32);
        if (m == 0) out[tile * 16 + i + 8 * half] = s + b2v;
    }
}

extern "C" void kernel_launch(void* const* d_in, const int* in_sizes, int n_in,
                              void* d_out, int out_size, void* d_ws, size_t ws_size,
                              hipStream_t stream) {
    const float* recordings       = (const float*)d_in[0];
    const float* sample_locations = (const float*)d_in[1];
    const float* emitter          = (const float*)d_in[2];
    const float* receivers        = (const float*)d_in[3];
    const float* W1               = (const float*)d_in[4];
    const float* b1               = (const float*)d_in[5];
    const float* W2               = (const float*)d_in[6];
    const float* b2               = (const float*)d_in[7];
    float* out                    = (float*)d_out;

    // 4096 row-tiles of 16 -> 512 blocks x 8 waves
    tof_wmma_kernel<<<512, 256, 0, stream>>>(
        recordings, sample_locations, emitter, receivers,
        W1, b1, W2, b2, out);
}
